// Net_12902081757444
// MI455X (gfx1250) — matrix operand order repaired and checked by reading, not compiled, not run on hardware
//
#include <hip/hip_runtime.h>
#include <hip/hip_bf16.h>
#include <math.h>

typedef _Float16 h16;
typedef __attribute__((ext_vector_type(16))) _Float16 v16h;
typedef __attribute__((ext_vector_type(8)))  float    v8f;

#define NNODES 100000
#define KNEI   8
#define NEDGES (NNODES * KNEI)
#define NGRAPH 128
#define NPADR  100096   /* NNODES rounded up to multiple of 128 */
#define EPSBN  1e-5f

static inline size_t align256(size_t x) { return (x + 255) & ~(size_t)255; }

// ---------------------------------------------------------------------------
// zero a float region (stats / accumulators must be zeroed every call)
// ---------------------------------------------------------------------------
__global__ void zero_f32(float* __restrict__ p, int n) {
    int i = blockIdx.x * blockDim.x + threadIdx.x;
    if (i < n) p[i] = 0.f;
}

// ---------------------------------------------------------------------------
// Edge message build + msg-MLP layer1 (9 -> 64), ReLU, f16 store, BN stats.
// Tiny K makes WMMA awkward here; VALU FMAs, 576 FMA/edge (~0.9 GFLOP total).
// ---------------------------------------------------------------------------
__global__ __launch_bounds__(256)
void msg_l1(const float* __restrict__ x, const int* __restrict__ src,
            const int* __restrict__ tgt, const float* __restrict__ ea,
            const float* __restrict__ W0, const float* __restrict__ b0,
            h16* __restrict__ h1, float* __restrict__ stats) {
    __shared__ float s_sum[64], s_sq[64];
    if (threadIdx.x < 64) { s_sum[threadIdx.x] = 0.f; s_sq[threadIdx.x] = 0.f; }
    __syncthreads();
    int e = blockIdx.x * blockDim.x + threadIdx.x;
    bool live = (e < NEDGES);
    int es = live ? e : 0;
    int s = src[es], t = tgt[es];
    float f[9];
#pragma unroll
    for (int i = 0; i < 4; ++i) {
        float xs = x[s * 4 + i];
        f[i] = xs;
        f[4 + i] = x[t * 4 + i] - xs;
    }
    f[8] = ea[es];
    for (int j = 0; j < 64; ++j) {
        float a = b0[j];
#pragma unroll
        for (int i = 0; i < 9; ++i) a = fmaf(f[i], W0[i * 64 + j], a);
        a = fmaxf(a, 0.f);
        if (live) {
            h1[(size_t)e * 64 + j] = (h16)a;
            atomicAdd(&s_sum[j], a);
            atomicAdd(&s_sq[j], a * a);
        }
    }
    __syncthreads();
    if (threadIdx.x < 64) {
        atomicAdd(&stats[threadIdx.x],      s_sum[threadIdx.x]);
        atomicAdd(&stats[64 + threadIdx.x], s_sq[threadIdx.x]);
    }
}

// ---------------------------------------------------------------------------
// Fold previous layer's BatchNorm affine into the next linear layer, emitting
// the weight TRANSPOSED (Wt[n][k], f16) so the GEMM can stage LDS tiles with
// contiguous 16B async DMAs:
//   a[k] = gamma[k]*rsqrt(var[k]+eps) ; c[k] = beta[k] - mean[k]*a[k]
//   Wt[n][k] = a[k]*W[k][n] ; b'[n] = bias[n] + sum_k c[k]*W[k][n]
// stats == nullptr -> identity fold. Rows [Nfull, Npad) are zero-padded.
// ---------------------------------------------------------------------------
__global__ void fold_bn(const float* __restrict__ stats, float invCount,
                        const float* __restrict__ gamma, const float* __restrict__ beta,
                        const float* __restrict__ W, const float* __restrict__ bias,
                        h16* __restrict__ Wt, float* __restrict__ bf,
                        int Kd, int Nfull, int Npad) {
    int n = blockIdx.x * blockDim.x + threadIdx.x;
    if (n >= Npad) return;
    if (n >= Nfull) {
        for (int k = 0; k < Kd; ++k) Wt[(size_t)n * Kd + k] = (h16)0.f;
        bf[n] = 0.f;
        return;
    }
    float bacc = bias[n];
    for (int k = 0; k < Kd; ++k) {
        float a = 1.f, c = 0.f;
        if (stats) {
            float m   = stats[k] * invCount;
            float var = stats[Kd + k] * invCount - m * m;
            a = gamma[k] * rsqrtf(var + EPSBN);
            c = beta[k] - m * a;
        }
        float w = W[(size_t)k * Nfull + n];
        Wt[(size_t)n * Kd + k] = (h16)(w * a);
        bacc += c * w;
    }
    bf[n] = bacc;
}

// ---------------------------------------------------------------------------
// Generic f16 WMMA GEMM:  C = [relu](A @ W + bf), optional BN-stat epilogue.
//   A  : f16 [Mpad x Kd] row-major (pad rows are zero)
//   Wt : f16 [Nw x Kd]   TRANSPOSED, BN-folded       (Kd must be mult. of 64)
//   C  : f16 [Mpad x Nw] (rows >= M stored as 0)
// Block = 256 threads = 8 waves; tile = 128 rows x 64 cols.
// K consumed in 64-wide super-chunks: the 64x64 f16 W tile (8KB) is staged by
// two paired global_load_async_to_lds_b128 DMAs per thread (ASYNCcnt-tracked;
// offset:16 advances BOTH the LDS and global address per the async-op spec),
// then 8 v_wmma_f32_16x16x32_f16 run per barrier pair.
// ---------------------------------------------------------------------------
__global__ __launch_bounds__(256)
void gemm_wmma(const h16* __restrict__ A, const h16* __restrict__ W,
               const float* __restrict__ bias, h16* __restrict__ C,
               float* __restrict__ stats, int M, int Kd, int Nw, int relu) {
    __shared__ __align__(16) h16 Wl[64 * 64];   // Wl[n*64 + k] (transposed tile)
    __shared__ float s_sum[64];
    __shared__ float s_sq[64];

    const int tid  = threadIdx.x;
    const int lane = tid & 31;
    const int wv   = tid >> 5;
    const int l15  = lane & 15;
    const int hs   = lane >> 4;                  // half-select (lanes 16-31)
    const int rowBase = blockIdx.x * 128 + wv * 16;
    const int colBase = blockIdx.y * 64;
    const bool doStats = (stats != nullptr);

    if (doStats && tid < 64) { s_sum[tid] = 0.f; s_sq[tid] = 0.f; }

    v8f acc[4] = {};
    const int row = rowBase + l15;
    const int nn  = tid >> 2;                    // 0..63 : tile row this thread stages
    const int sub = tid & 3;                     // 0..3  : 16-half chunk within row
    const unsigned ldsa =
        (unsigned)(size_t)(&Wl[nn * 64 + sub * 16]);  // LDS byte offset (low 32b)

    for (int k0 = 0; k0 < Kd; k0 += 64) {
        // async DMA stage of Wt[colBase..+63][k0..k0+63] -> Wl (32B per thread)
        {
            const h16* gsrc = W + (size_t)(colBase + nn) * Kd + k0 + sub * 16;
            asm volatile("global_load_async_to_lds_b128 %0, %1, off"
                         :: "v"(ldsa), "v"(gsrc) : "memory");
            asm volatile("global_load_async_to_lds_b128 %0, %1, off offset:16"
                         :: "v"(ldsa), "v"(gsrc) : "memory");
        }

        // A fragments for both 32-wide sub-chunks, per ISA 16-bit A layout:
        // lanes 0-15 hold K {0..7, 16..23}, lanes 16-31 hold K {8..15, 24..31}
        const h16* ap = A + (size_t)row * Kd + k0 + hs * 8;
        if (k0 + 64 < Kd) __builtin_prefetch(ap + 64, 0, 1);
        v16h af0, af1;
#pragma unroll
        for (int i = 0; i < 8; ++i) af0[i]     = ap[i];
#pragma unroll
        for (int i = 0; i < 8; ++i) af0[8 + i] = ap[16 + i];
#pragma unroll
        for (int i = 0; i < 8; ++i) af1[i]     = ap[32 + i];
#pragma unroll
        for (int i = 0; i < 8; ++i) af1[8 + i] = ap[48 + i];

        asm volatile("s_wait_asynccnt 0x0" ::: "memory");
        __syncthreads();

        // 8 WMMAs per stage: 4 column tiles x 2 K sub-chunks
#pragma unroll
        for (int n = 0; n < 4; ++n) {
            const h16* bp = &Wl[(n * 16 + l15) * 64 + hs * 8];
            v16h bf0, bf1;
#pragma unroll
            for (int i = 0; i < 16; ++i) bf0[i] = bp[i + (i & 8)];
#pragma unroll
            for (int i = 0; i < 16; ++i) bf1[i] = bp[32 + i + (i & 8)];
            acc[n] = __builtin_amdgcn_wmma_f32_16x16x32_f16(
                false, af0, false, bf0, (short)0, acc[n], false, false);
            acc[n] = __builtin_amdgcn_wmma_f32_16x16x32_f16(
                false, af1, false, bf1, (short)0, acc[n], false, false);
        }
        __syncthreads();
    }

    // epilogue: bias, relu, f16 store (pad rows -> 0), BN stats
#pragma unroll
    for (int n = 0; n < 4; ++n) {
        int col = colBase + n * 16 + l15;
        float bv = bias[col];
        float lsum = 0.f, lsq = 0.f;
#pragma unroll
        for (int j = 0; j < 8; ++j) {
            int r = rowBase + hs * 8 + j;        // C/D layout: lanes16-31 -> M+8
            float val = acc[n][j] + bv;
            if (relu) val = fmaxf(val, 0.f);
            bool live = (r < M);
            C[(size_t)r * Nw + col] = (h16)(live ? val : 0.f);
            if (live) { lsum += val; lsq += val * val; }
        }
        if (doStats) {
            atomicAdd(&s_sum[n * 16 + l15], lsum);
            atomicAdd(&s_sq[n * 16 + l15], lsq);
        }
    }
    if (doStats) {
        __syncthreads();
        if (tid < 64) {
            atomicAdd(&stats[colBase + tid],      s_sum[tid]);
            atomicAdd(&stats[Nw + colBase + tid], s_sq[tid]);
        }
    }
}

// ---------------------------------------------------------------------------
// Per-(node, head) softmax attention, already averaged over the K=8 queries
// (mean over K commutes with the later @wo GEMM). Pad rows write zeros.
// ---------------------------------------------------------------------------
__global__ __launch_bounds__(256)
void attn_mean(const h16* __restrict__ q, const h16* __restrict__ k,
               const h16* __restrict__ v, h16* __restrict__ obar, int npadRows) {
    int gid = blockIdx.x * blockDim.x + threadIdx.x;
    int node = gid >> 3, head = gid & 7;
    if (node >= npadRows) return;
    if (node >= NNODES) {
#pragma unroll
        for (int d = 0; d < 8; ++d) obar[(size_t)node * 64 + head * 8 + d] = (h16)0.f;
        return;
    }
    size_t base = (size_t)node * 8 * 64 + (size_t)head * 8;
    float kk[8][8], vv[8][8];
#pragma unroll
    for (int r = 0; r < 8; ++r)
#pragma unroll
        for (int d = 0; d < 8; ++d) {
            kk[r][d] = (float)k[base + r * 64 + d];
            vv[r][d] = (float)v[base + r * 64 + d];
        }
    float acc[8] = {0.f, 0.f, 0.f, 0.f, 0.f, 0.f, 0.f, 0.f};
    const float scale = 0.35355339059327373f;     // 1/sqrt(8)
    for (int qi = 0; qi < 8; ++qi) {
        float qr[8];
#pragma unroll
        for (int d = 0; d < 8; ++d) qr[d] = (float)q[base + qi * 64 + d];
        float s[8], m = -1e30f;
#pragma unroll
        for (int ki = 0; ki < 8; ++ki) {
            float dot = 0.f;
#pragma unroll
            for (int d = 0; d < 8; ++d) dot = fmaf(qr[d], kk[ki][d], dot);
            s[ki] = dot * scale;
            m = fmaxf(m, s[ki]);
        }
        float ssum = 0.f;
#pragma unroll
        for (int ki = 0; ki < 8; ++ki) { s[ki] = __expf(s[ki] - m); ssum += s[ki]; }
        float inv = 1.f / ssum;
#pragma unroll
        for (int ki = 0; ki < 8; ++ki) {
            float w = s[ki] * inv;
#pragma unroll
            for (int d = 0; d < 8; ++d) acc[d] = fmaf(w, vv[ki][d], acc[d]);
        }
    }
#pragma unroll
    for (int d = 0; d < 8; ++d)
        obar[(size_t)node * 64 + head * 8 + d] = (h16)(acc[d] * 0.125f);
}

// ---------------------------------------------------------------------------
// Graph segment-mean exploiting SORTED batch ids: each block owns a 64-node
// chunk, each thread 4 contiguous columns (coalesced 2KB row reads).
// Partial sums are kept in registers across a run of equal batch ids and
// flushed with one global atomic per boundary (up to 64x fewer atomics than
// per-element atomics).
// ---------------------------------------------------------------------------
__global__ __launch_bounds__(256)
void segsum_sorted(const h16* __restrict__ h, const int* __restrict__ batch,
                   float* __restrict__ gsum, float* __restrict__ gcnt) {
    int n0  = blockIdx.x * 64;
    int col = threadIdx.x * 4;
    float s0 = 0.f, s1 = 0.f, s2 = 0.f, s3 = 0.f;
    int cur = -1;
    for (int i = 0; i < 64; ++i) {
        int node = n0 + i;
        if (node >= NNODES) break;
        int b = batch[node];
        if (b != cur) {
            if (cur >= 0) {
                atomicAdd(&gsum[(size_t)cur * 1024 + col + 0], s0);
                atomicAdd(&gsum[(size_t)cur * 1024 + col + 1], s1);
                atomicAdd(&gsum[(size_t)cur * 1024 + col + 2], s2);
                atomicAdd(&gsum[(size_t)cur * 1024 + col + 3], s3);
            }
            cur = b; s0 = s1 = s2 = s3 = 0.f;
        }
        const h16* hp = h + (size_t)node * 1024 + col;
        s0 += (float)hp[0]; s1 += (float)hp[1];
        s2 += (float)hp[2]; s3 += (float)hp[3];
    }
    if (cur >= 0) {
        atomicAdd(&gsum[(size_t)cur * 1024 + col + 0], s0);
        atomicAdd(&gsum[(size_t)cur * 1024 + col + 1], s1);
        atomicAdd(&gsum[(size_t)cur * 1024 + col + 2], s2);
        atomicAdd(&gsum[(size_t)cur * 1024 + col + 3], s3);
    }
    if (threadIdx.x == 0) {     // run-length node counts
        int c = -1; float run = 0.f;
        for (int i = 0; i < 64; ++i) {
            int node = n0 + i;
            if (node >= NNODES) break;
            int b = batch[node];
            if (b != c) {
                if (c >= 0) atomicAdd(&gcnt[c], run);
                c = b; run = 0.f;
            }
            run += 1.f;
        }
        if (c >= 0) atomicAdd(&gcnt[c], run);
    }
}

__global__ void gdiv(const float* __restrict__ gsum, const float* __restrict__ gcnt,
                     h16* __restrict__ gbar) {
    int i = blockIdx.x * blockDim.x + threadIdx.x;
    if (i >= NGRAPH * 1024) return;
    float c = fmaxf(gcnt[i >> 10], 1.0f);
    gbar[i] = (h16)(gsum[i] / c);
}

// ---------------------------------------------------------------------------
// log-softmax over the 40 real columns of the padded [128 x 64] f16 output
// ---------------------------------------------------------------------------
__global__ void log_softmax_k(const h16* __restrict__ g, float* __restrict__ out) {
    int r = blockIdx.x * blockDim.x + threadIdx.x;
    if (r >= NGRAPH) return;
    float vals[40], m = -1e30f;
    for (int j = 0; j < 40; ++j) { vals[j] = (float)g[r * 64 + j]; m = fmaxf(m, vals[j]); }
    float s = 0.f;
    for (int j = 0; j < 40; ++j) s += __expf(vals[j] - m);
    float lse = m + __logf(s);
    for (int j = 0; j < 40; ++j) out[r * 40 + j] = vals[j] - lse;
}

// ---------------------------------------------------------------------------
extern "C" void kernel_launch(void* const* d_in, const int* in_sizes, int n_in,
                              void* d_out, int out_size, void* d_ws, size_t ws_size,
                              hipStream_t stream) {
    (void)in_sizes; (void)n_in; (void)out_size; (void)ws_size;
    // inputs in setup_inputs() insertion order
    const float* x    = (const float*)d_in[0];
    const int*   src  = (const int*)d_in[1];
    const int*   tgt  = (const int*)d_in[1] + NEDGES;
    const float* ea   = (const float*)d_in[2];
    const int*   bat  = (const int*)d_in[3];
    const float *w0=(const float*)d_in[4],  *b0=(const float*)d_in[5],
                *g0=(const float*)d_in[6],  *be0=(const float*)d_in[7];
    const float *w1=(const float*)d_in[8],  *b1=(const float*)d_in[9],
                *g1p=(const float*)d_in[10],*be1=(const float*)d_in[11];
    const float *w2=(const float*)d_in[12], *b2=(const float*)d_in[13],
                *g2p=(const float*)d_in[14],*be2=(const float*)d_in[15];
    const float *wq=(const float*)d_in[16], *bq=(const float*)d_in[17];
    const float *wk=(const float*)d_in[18], *bk=(const float*)d_in[19];
    const float *wv=(const float*)d_in[20], *bv=(const float*)d_in[21];
    const float *wo=(const float*)d_in[22], *bo=(const float*)d_in[23];
    const float *wl1=(const float*)d_in[24],*bl1=(const float*)d_in[25],
                *gl1=(const float*)d_in[26],*bel1=(const float*)d_in[27];
    const float *wm1=(const float*)d_in[28],*bm1=(const float*)d_in[29],
                *gm1=(const float*)d_in[30],*bem1=(const float*)d_in[31];
    const float *wm2=(const float*)d_in[32],*bm2=(const float*)d_in[33],
                *gm2=(const float*)d_in[34],*bem2=(const float*)d_in[35];
    const float *wou=(const float*)d_in[36],*bou=(const float*)d_in[37];

    // ---- workspace bump allocator (with deliberate aliasing) ----
    char* ws = (char*)d_ws;
    const size_t S = (size_t)NEDGES * 64 * 2;          // [E,64] f16 bytes
    h16* h1 = (h16*)(ws + 0);                          // later reused as q
    h16* h2 = (h16*)(ws + S);                          // later reused as k
    h16* h3 = (h16*)(ws + 2 * S);
    h16* vb = (h16*)(ws + 3 * S);
    h16* qb = h1;
    h16* kb = h2;
    h16* hL = h3;   // [NPADR,1024] f16, overlays h3+vb region (both dead by then)
    size_t cur = 2 * S + align256((size_t)NPADR * 1024 * 2);
    auto alloc = [&](size_t bytes) -> char* {
        char* p = ws + cur; cur = align256(cur + bytes); return p;
    };
    h16* obar = (h16*)alloc((size_t)NPADR * 64 * 2);
    h16* x1   = (h16*)alloc((size_t)NPADR * 64 * 2);
    // stats block (zeroed every call)
    const int ZF = 128 * 3 + 2048 + 1024 + 512 + 131072 + 128;
    float* st  = (float*)alloc((size_t)ZF * 4);
    float *st1 = st, *st2 = st1 + 128, *st3 = st2 + 128, *stL = st3 + 128;
    float *stM1 = stL + 2048, *stM2 = stM1 + 1024;
    float *gsum = stM2 + 512, *gcnt = gsum + 131072;
    // folded biases (f32)
    float* b1f  = (float*)alloc(64 * 4);   float* b2f  = (float*)alloc(64 * 4);
    float* bqf  = (float*)alloc(64 * 4);   float* bkf  = (float*)alloc(64 * 4);
    float* bvf  = (float*)alloc(64 * 4);   float* bof  = (float*)alloc(64 * 4);
    float* bl1f = (float*)alloc(1024 * 4); float* bm1f = (float*)alloc(512 * 4);
    float* bm2f = (float*)alloc(256 * 4);  float* bouf = (float*)alloc(64 * 4);
    // folded f16 weights (transposed [Npad x Kd])
    h16* W1f  = (h16*)alloc(4096 * 2);     h16* W2f  = (h16*)alloc(4096 * 2);
    h16* Wqf  = (h16*)alloc(4096 * 2);     h16* Wkf  = (h16*)alloc(4096 * 2);
    h16* Wvf  = (h16*)alloc(4096 * 2);     h16* Wof  = (h16*)alloc(4096 * 2);
    h16* Wl1f = (h16*)alloc(65536 * 2);    h16* Wm1f = (h16*)alloc(524288 * 2);
    h16* Wm2f = (h16*)alloc(131072 * 2);   h16* Wouf = (h16*)alloc(16384 * 2);
    // small activations
    h16* gbar = (h16*)alloc(NGRAPH * 1024 * 2);
    h16* gg1  = (h16*)alloc(NGRAPH * 512 * 2);
    h16* gg2  = (h16*)alloc(NGRAPH * 256 * 2);
    h16* gout = (h16*)alloc(NGRAPH * 64 * 2);

    const dim3 blk(256);
    const int EB = NEDGES / 128;   // 6250 row-blocks over edges
    const int NB = NPADR / 128;    // 782 row-blocks over (padded) nodes
    const float invE = 1.f / (float)NEDGES;
    const float invN = 1.f / (float)NNODES;
    const float invB = 1.f / (float)NGRAPH;

    zero_f32<<<(ZF + 255) / 256, 256, 0, stream>>>(st, ZF);
    // msg layer1 + stats1
    msg_l1<<<(NEDGES + 255) / 256, 256, 0, stream>>>(x, src, tgt, ea, w0, b0, h1, st1);
    // layer2: fold BN1 into w1, GEMM h1->h2 (+stats2)
    fold_bn<<<1, 64, 0, stream>>>(st1, invE, g0, be0, w1, b1, W1f, b1f, 64, 64, 64);
    gemm_wmma<<<dim3(EB, 1), blk, 0, stream>>>(h1, W1f, b1f, h2, st2, NEDGES, 64, 64, 1);
    // layer3: fold BN2 into w2, GEMM h2->h3 (+stats3)
    fold_bn<<<1, 64, 0, stream>>>(st2, invE, g1p, be1, w2, b2, W2f, b2f, 64, 64, 64);
    gemm_wmma<<<dim3(EB, 1), blk, 0, stream>>>(h2, W2f, b2f, h3, st3, NEDGES, 64, 64, 1);
    // QKV: fold BN3 into each projection
    fold_bn<<<1, 64, 0, stream>>>(st3, invE, g2p, be2, wq, bq, Wqf, bqf, 64, 64, 64);
    fold_bn<<<1, 64, 0, stream>>>(st3, invE, g2p, be2, wk, bk, Wkf, bkf, 64, 64, 64);
    fold_bn<<<1, 64, 0, stream>>>(st3, invE, g2p, be2, wv, bv, Wvf, bvf, 64, 64, 64);
    gemm_wmma<<<dim3(EB, 1), blk, 0, stream>>>(h3, Wqf, bqf, qb, nullptr, NEDGES, 64, 64, 0);
    gemm_wmma<<<dim3(EB, 1), blk, 0, stream>>>(h3, Wkf, bkf, kb, nullptr, NEDGES, 64, 64, 0);
    gemm_wmma<<<dim3(EB, 1), blk, 0, stream>>>(h3, Wvf, bvf, vb, nullptr, NEDGES, 64, 64, 0);
    // attention, averaged over queries (== scatter-mean over tgt, pre-wo)
    attn_mean<<<(NPADR * 8 + 255) / 256, 256, 0, stream>>>(qb, kb, vb, obar, NPADR);
    // x1 = obar @ wo + bo
    fold_bn<<<1, 64, 0, stream>>>(nullptr, 0.f, b0, b0, wo, bo, Wof, bof, 64, 64, 64);
    gemm_wmma<<<dim3(NB, 1), blk, 0, stream>>>(obar, Wof, bof, x1, nullptr, NNODES, 64, 64, 0);
    // lin1: 64 -> 1024 (+statsL over N rows)
    fold_bn<<<16, 64, 0, stream>>>(nullptr, 0.f, b0, b0, wl1, bl1, Wl1f, bl1f, 64, 1024, 1024);
    gemm_wmma<<<dim3(NB, 16), blk, 0, stream>>>(x1, Wl1f, bl1f, hL, stL, NNODES, 64, 1024, 1);
    // graph segment-mean (BN of lin1 commutes with the mean; folded next)
    segsum_sorted<<<(NNODES + 63) / 64, 256, 0, stream>>>(hL, bat, gsum, gcnt);
    gdiv<<<(NGRAPH * 1024 + 255) / 256, 256, 0, stream>>>(gsum, gcnt, gbar);
    // mlp1: fold lin1-BN, GEMM 128x1024x512 (+statsM1)
    fold_bn<<<8, 64, 0, stream>>>(stL, invN, gl1, bel1, wm1, bm1, Wm1f, bm1f, 1024, 512, 512);
    gemm_wmma<<<dim3(1, 8), blk, 0, stream>>>(gbar, Wm1f, bm1f, gg1, stM1, NGRAPH, 1024, 512, 1);
    // mlp2: fold mlp1-BN, GEMM 128x512x256 (+statsM2)
    fold_bn<<<4, 64, 0, stream>>>(stM1, invB, gm1, bem1, wm2, bm2, Wm2f, bm2f, 512, 256, 256);
    gemm_wmma<<<dim3(1, 4), blk, 0, stream>>>(gg1, Wm2f, bm2f, gg2, stM2, NGRAPH, 512, 256, 1);
    // lin_out: fold mlp2-BN, GEMM 128x256x40 (padded to 64 cols)
    fold_bn<<<1, 64, 0, stream>>>(stM2, invB, gm2, bem2, wou, bou, Wouf, bouf, 256, 40, 64);
    gemm_wmma<<<dim3(1, 1), blk, 0, stream>>>(gg2, Wouf, bouf, gout, nullptr, NGRAPH, 256, 64, 0);
    // log-softmax -> d_out [128 x 40] f32
    log_softmax_k<<<1, 128, 0, stream>>>(gout, (float*)d_out);
}
